// TransCube_71270687310406
// MI455X (gfx1250) — matrix operand order repaired and checked by reading, not compiled, not run on hardware
//
#include <hip/hip_runtime.h>
#include <math.h>

// ---------------------------------------------------------------------------
// MI455X (gfx1250) TransCube attention block.
//   - bf16 WMMA (v_wmma_f32_16x16x32_bf16) for QKV GEMM, QK^T, PV, out-proj
//   - per-(seq,head) fused attention: scores/softmax/probs never leave LDS
//   - memory-bound problem (~600MB @ 23.3TB/s); minimize HBM round trips
// ---------------------------------------------------------------------------

typedef __attribute__((ext_vector_type(16))) __bf16 v16bf;
typedef __attribute__((ext_vector_type(8)))  float  v8f;

union Frag { uint4 u[2]; v16bf v; };

__device__ __forceinline__ unsigned short f2bf(float f) {
  unsigned int u = __float_as_uint(f);
  u += 0x7FFFu + ((u >> 16) & 1u);       // round-to-nearest-even truncation
  return (unsigned short)(u >> 16);
}

__device__ __forceinline__ v8f wmma_bf16(const Frag& a, const Frag& b, v8f c) {
  // 8-arg form: (neg_a, A, neg_b, B, c_mod, C, reuse_a, reuse_b)
  return __builtin_amdgcn_wmma_f32_16x16x32_bf16(false, a.v, false, b.v,
                                                 (short)0, c, false, false);
}

// ---------------------------------------------------------------------------
// Kernel 1: QKV projection.  out[n,o] = sum_c X[n,c] * W[o,c]
// grid = (576 sequences, 6)  y/2 selects {Wq,Wk,Wv}, y&1 selects 128-col half.
// Block tile: M=128 tokens (one cube sequence, gathered), N=128, K=256.
// A-tile and W-tile staged in LDS as bf16 (stride 264 halves = 528B, 16B mult).
// ---------------------------------------------------------------------------
__global__ void tc_qkv(const float* __restrict__ x,
                       const float* __restrict__ wq,
                       const float* __restrict__ wk,
                       const float* __restrict__ wv,
                       float* __restrict__ q, float* __restrict__ k,
                       float* __restrict__ v) {
  extern __shared__ char dynsm[];
  unsigned short* A = (unsigned short*)dynsm;   // [128][264]
  unsigned short* W = A + 128 * 264;            // [128][264]
  const int seq  = blockIdx.x;
  const int ysel = blockIdx.y;
  const int tid  = threadIdx.x;

  // gather one cube sequence: seq -> (d, hb, g); token t -> (wb,hc,wc)
  {
    const int d   = seq / 72;
    const int rem = seq - d * 72;
    const int hb  = rem / 6;
    const int g   = rem - hb * 6;
    const int t   = tid >> 1;
    const int ch0 = (tid & 1) * 128;
    const int wb  = g * 2 + (t >> 6);
    const int hh  = hb * 8 + ((t >> 3) & 7);
    const int ww  = wb * 8 + (t & 7);
    const float* src = x + (size_t)((d * 96 + hh) * 96 + ww) * 256 + ch0;
    unsigned short* dst = A + t * 264 + ch0;
#pragma unroll
    for (int i = 0; i < 32; ++i) {
      float4 f = ((const float4*)src)[i];
      dst[i * 4 + 0] = f2bf(f.x); dst[i * 4 + 1] = f2bf(f.y);
      dst[i * 4 + 2] = f2bf(f.z); dst[i * 4 + 3] = f2bf(f.w);
    }
  }
  // stage 128 weight rows (output channels), full K=256, as bf16
  {
    const float* wptr = (ysel < 2) ? wq : (ysel < 4) ? wk : wv;
    const int n0  = (ysel & 1) * 128;
    const int r   = tid >> 1;
    const int ch0 = (tid & 1) * 128;
    const float* src = wptr + (size_t)(n0 + r) * 256 + ch0;
    __builtin_prefetch(src + 256, 0, 1);   // global_prefetch_b8, next row
    unsigned short* dst = W + r * 264 + ch0;
#pragma unroll
    for (int i = 0; i < 32; ++i) {
      float4 f = ((const float4*)src)[i];
      dst[i * 4 + 0] = f2bf(f.x); dst[i * 4 + 1] = f2bf(f.y);
      dst[i * 4 + 2] = f2bf(f.z); dst[i * 4 + 3] = f2bf(f.w);
    }
  }
  __syncthreads();

  const int wave = tid >> 5, lane = tid & 31;
  const int m0 = wave * 16, lm = lane & 15, hs = lane >> 4;
  v8f acc[8] = {};
#pragma unroll
  for (int kc = 0; kc < 8; ++kc) {
    // A 16x32 bf16: lanes0-15 hold K{hs*8..+8, 16+hs*8..+8} per ISA layout
    Frag a;
    const unsigned short* ap = A + (m0 + lm) * 264 + kc * 32 + hs * 8;
    a.u[0] = *(const uint4*)(ap);
    a.u[1] = *(const uint4*)(ap + 16);
#pragma unroll
    for (int j = 0; j < 8; ++j) {
      // B 32x16: lane = out column, holds K{hs*8..+8, 16+hs*8..+8} of W row
      Frag b;
      const unsigned short* bp = W + (j * 16 + lm) * 264 + kc * 32 + hs * 8;
      b.u[0] = *(const uint4*)(bp);
      b.u[1] = *(const uint4*)(bp + 16);
      acc[j] = wmma_bf16(a, b, acc[j]);
    }
  }
  float* dstm = (ysel < 2) ? q : (ysel < 4) ? k : v;
  const int col0 = (ysel & 1) * 128;
#pragma unroll
  for (int j = 0; j < 8; ++j)
#pragma unroll
    for (int r = 0; r < 8; ++r)
      dstm[(size_t)(seq * 128 + m0 + hs * 8 + r) * 256 + col0 + j * 16 + lm] =
          acc[j][r];
}

// ---------------------------------------------------------------------------
// Kernel 2: fused attention per (seq, head).  q,k normalized over token axis.
// LDS: qn/kn bf16 [128][40], v^T bf16 [32][136], S f32 [128][132],
//      P bf16 [128][136]  => 131584 B dynamic (under 320KB WGP budget).
// ---------------------------------------------------------------------------
__global__ void tc_attn(const float* __restrict__ q, const float* __restrict__ kk,
                        const float* __restrict__ v, const float* __restrict__ rescale,
                        float* __restrict__ o) {
  extern __shared__ char dynsm[];
  unsigned short* QN = (unsigned short*)dynsm;     // 128*40 halves
  unsigned short* KN = QN + 128 * 40;
  unsigned short* VT = KN + 128 * 40;              // 32*136 halves (d-major)
  float* S          = (float*)(dynsm + 29184);     // 128*132 f32
  unsigned short* P = (unsigned short*)(dynsm + 29184 + 67584); // 128*136
  __shared__ float partq[8][32], partk[8][32], red[2][32];

  const int seq = blockIdx.x, hh = blockIdx.y;
  const int tid = threadIdx.x;
  const size_t base = (size_t)seq * 128 * 256 + hh * 32;

  // pass 1: per-channel sum of squares over 128 tokens (dim=-2 normalize)
  {
    const int ch = tid & 31, rg = tid >> 5;
    float sq = 0.f, sk = 0.f;
    for (int r = rg; r < 128; r += 8) {
      float qv = q[base + (size_t)r * 256 + ch];
      float kv = kk[base + (size_t)r * 256 + ch];
      sq += qv * qv; sk += kv * kv;
    }
    partq[rg][ch] = sq; partk[rg][ch] = sk;
  }
  __syncthreads();
  if (tid < 32) {
    float s = 0.f;
    for (int i = 0; i < 8; ++i) s += partq[i][tid];
    red[0][tid] = 1.f / fmaxf(sqrtf(s), 1e-12f);
  } else if (tid < 64) {
    const int c = tid - 32;
    float s = 0.f;
    for (int i = 0; i < 8; ++i) s += partk[i][c];
    red[1][c] = 1.f / fmaxf(sqrtf(s), 1e-12f);
  }
  __syncthreads();
  // pass 2: normalized bf16 q/k tiles + v transposed to d-major
  {
    const int r = tid >> 1, half = tid & 1;
#pragma unroll
    for (int j = 0; j < 16; ++j) {
      const int ch = half * 16 + j;
      const size_t gi = base + (size_t)r * 256 + ch;
      QN[r * 40 + ch]  = f2bf(q[gi] * red[0][ch]);
      KN[r * 40 + ch]  = f2bf(kk[gi] * red[1][ch]);
      VT[ch * 136 + r] = f2bf(v[gi]);
    }
  }
  __syncthreads();

  const int wave = tid >> 5, lane = tid & 31;
  const int lm = lane & 15, hs = lane >> 4;
  const int m0 = wave * 16;
  const float rs = rescale[hh];

  // S = qn @ kn^T * rescale  (K = 32 = head dim, single WMMA per 16x16 tile)
  {
    Frag a;
    const unsigned short* ap = QN + (m0 + lm) * 40 + hs * 8;
    a.u[0] = *(const uint4*)(ap);
    a.u[1] = *(const uint4*)(ap + 16);
#pragma unroll
    for (int j = 0; j < 8; ++j) {
      Frag b;
      const unsigned short* bp = KN + (j * 16 + lm) * 40 + hs * 8;
      b.u[0] = *(const uint4*)(bp);
      b.u[1] = *(const uint4*)(bp + 16);
      v8f sac = {};
      sac = wmma_bf16(a, b, sac);
#pragma unroll
      for (int r = 0; r < 8; ++r)
        S[(m0 + hs * 8 + r) * 132 + j * 16 + lm] = sac[r] * rs;
    }
  }
  __syncthreads();

  // row softmax over 128 cols; each wave owns its 16 rows; lanes hold 4 cols
#pragma unroll 1
  for (int i = 0; i < 16; ++i) {
    const int row = m0 + i;
    const float4 xv = *(const float4*)&S[row * 132 + (lane << 2)];
    float m = fmaxf(fmaxf(xv.x, xv.y), fmaxf(xv.z, xv.w));
    for (int off = 16; off > 0; off >>= 1) m = fmaxf(m, __shfl_xor(m, off));
    float e0 = expf(xv.x - m), e1 = expf(xv.y - m);
    float e2 = expf(xv.z - m), e3 = expf(xv.w - m);
    float s = e0 + e1 + e2 + e3;
    for (int off = 16; off > 0; off >>= 1) s += __shfl_xor(s, off);
    const float inv = 1.f / s;
    unsigned short* pp = P + row * 136 + (lane << 2);
    pp[0] = f2bf(e0 * inv); pp[1] = f2bf(e1 * inv);
    pp[2] = f2bf(e2 * inv); pp[3] = f2bf(e3 * inv);
  }
  __syncthreads();

  // O = P @ v  (K = 128 tokens, 4 WMMA steps; B tile from d-major v^T)
  v8f oacc[2] = {};
#pragma unroll
  for (int kc = 0; kc < 4; ++kc) {
    Frag a;
    const unsigned short* ap = P + (m0 + lm) * 136 + kc * 32 + hs * 8;
    a.u[0] = *(const uint4*)(ap);
    a.u[1] = *(const uint4*)(ap + 16);
#pragma unroll
    for (int j = 0; j < 2; ++j) {
      Frag b;
      const unsigned short* bp = VT + (j * 16 + lm) * 136 + kc * 32 + hs * 8;
      b.u[0] = *(const uint4*)(bp);
      b.u[1] = *(const uint4*)(bp + 16);
      oacc[j] = wmma_bf16(a, b, oacc[j]);
    }
  }
#pragma unroll
  for (int j = 0; j < 2; ++j)
#pragma unroll
    for (int r = 0; r < 8; ++r)
      o[(size_t)(seq * 128 + m0 + hs * 8 + r) * 256 + hh * 32 + j * 16 + lm] =
          oacc[j][r];
}

// ---------------------------------------------------------------------------
// Kernel 3: output projection + bias: out[n,c] = sum_o o[n,o]*Pw[c,o] + b[c]
// grid = (576 row tiles, 2 column halves); same WMMA core as tc_qkv.
// Writes out_c in flat token order == spatial (d,h,w) order per reference.
// ---------------------------------------------------------------------------
__global__ void tc_proj(const float* __restrict__ om, const float* __restrict__ pw,
                        const float* __restrict__ pb, float* __restrict__ out) {
  extern __shared__ char dynsm[];
  unsigned short* A = (unsigned short*)dynsm;   // [128][264]
  unsigned short* W = A + 128 * 264;
  const int mblk = blockIdx.x;
  const int n0   = blockIdx.y * 128;
  const int tid  = threadIdx.x;
  {
    const int r = tid >> 1, ch0 = (tid & 1) * 128;
    const float* src = om + (size_t)(mblk * 128 + r) * 256 + ch0;
    unsigned short* dst = A + r * 264 + ch0;
#pragma unroll
    for (int i = 0; i < 32; ++i) {
      float4 f = ((const float4*)src)[i];
      dst[i * 4 + 0] = f2bf(f.x); dst[i * 4 + 1] = f2bf(f.y);
      dst[i * 4 + 2] = f2bf(f.z); dst[i * 4 + 3] = f2bf(f.w);
    }
  }
  {
    const int r = tid >> 1, ch0 = (tid & 1) * 128;
    const float* src = pw + (size_t)(n0 + r) * 256 + ch0;
    unsigned short* dst = W + r * 264 + ch0;
#pragma unroll
    for (int i = 0; i < 32; ++i) {
      float4 f = ((const float4*)src)[i];
      dst[i * 4 + 0] = f2bf(f.x); dst[i * 4 + 1] = f2bf(f.y);
      dst[i * 4 + 2] = f2bf(f.z); dst[i * 4 + 3] = f2bf(f.w);
    }
  }
  __syncthreads();

  const int wave = tid >> 5, lane = tid & 31;
  const int m0 = wave * 16, lm = lane & 15, hs = lane >> 4;
  v8f acc[8] = {};
#pragma unroll
  for (int kc = 0; kc < 8; ++kc) {
    Frag a;
    const unsigned short* ap = A + (m0 + lm) * 264 + kc * 32 + hs * 8;
    a.u[0] = *(const uint4*)(ap);
    a.u[1] = *(const uint4*)(ap + 16);
#pragma unroll
    for (int j = 0; j < 8; ++j) {
      Frag b;
      const unsigned short* bp = W + (j * 16 + lm) * 264 + kc * 32 + hs * 8;
      b.u[0] = *(const uint4*)(bp);
      b.u[1] = *(const uint4*)(bp + 16);
      acc[j] = wmma_bf16(a, b, acc[j]);
    }
  }
#pragma unroll
  for (int j = 0; j < 8; ++j) {
    const float bias = pb[n0 + j * 16 + lm];
#pragma unroll
    for (int r = 0; r < 8; ++r)
      out[(size_t)(mblk * 128 + m0 + hs * 8 + r) * 256 + n0 + j * 16 + lm] =
          acc[j][r] + bias;
  }
}

// ---------------------------------------------------------------------------
// Kernel 4: depthwise 3x3x3 conv over [8,96,96,256] channel-last (SAME pad).
// phase 0: dst = gelu_exact(conv(src, wt));  phase 1: addout += conv(src, wt)
// ---------------------------------------------------------------------------
__global__ void tc_dwconv(const float* __restrict__ src, const float* __restrict__ wt,
                          float* __restrict__ dst, float* __restrict__ addout,
                          int phase) {
  const int tid = threadIdx.x;
  const int p   = blockIdx.x * 4 + (tid >> 6);  // flat spatial point
  const int c0  = (tid & 63) * 4;
  const int d   = p / 9216;
  const int hw  = p - d * 9216;
  const int h   = hw / 96;
  const int w   = hw - h * 96;
  float a0 = 0.f, a1 = 0.f, a2 = 0.f, a3 = 0.f;
#pragma unroll
  for (int dz = -1; dz <= 1; ++dz) {
    const int zz = d + dz; if ((unsigned)zz >= 8u) continue;
#pragma unroll
    for (int dy = -1; dy <= 1; ++dy) {
      const int yy = h + dy; if ((unsigned)yy >= 96u) continue;
#pragma unroll
      for (int dx = -1; dx <= 1; ++dx) {
        const int xx = w + dx; if ((unsigned)xx >= 96u) continue;
        const int tap = (dz + 1) * 9 + (dy + 1) * 3 + (dx + 1);
        const float4 xv =
            *(const float4*)(src + (size_t)((zz * 96 + yy) * 96 + xx) * 256 + c0);
        a0 += xv.x * wt[(c0 + 0) * 27 + tap];
        a1 += xv.y * wt[(c0 + 1) * 27 + tap];
        a2 += xv.z * wt[(c0 + 2) * 27 + tap];
        a3 += xv.w * wt[(c0 + 3) * 27 + tap];
      }
    }
  }
  const size_t oi = (size_t)p * 256 + c0;
  if (phase == 0) {  // exact GELU: 0.5*x*(1+erf(x/sqrt(2)))
    dst[oi + 0] = 0.5f * a0 * (1.f + erff(a0 * 0.70710678118654752f));
    dst[oi + 1] = 0.5f * a1 * (1.f + erff(a1 * 0.70710678118654752f));
    dst[oi + 2] = 0.5f * a2 * (1.f + erff(a2 * 0.70710678118654752f));
    dst[oi + 3] = 0.5f * a3 * (1.f + erff(a3 * 0.70710678118654752f));
  } else {
    addout[oi + 0] += a0; addout[oi + 1] += a1;
    addout[oi + 2] += a2; addout[oi + 3] += a3;
  }
}

// ---------------------------------------------------------------------------
extern "C" void kernel_launch(void* const* d_in, const int* in_sizes, int n_in,
                              void* d_out, int out_size, void* d_ws, size_t ws_size,
                              hipStream_t stream) {
  const float* x       = (const float*)d_in[0];
  const float* wq      = (const float*)d_in[1];
  const float* wk      = (const float*)d_in[2];
  const float* wv      = (const float*)d_in[3];
  const float* rescale = (const float*)d_in[4];
  const float* proj_w  = (const float*)d_in[5];
  const float* proj_b  = (const float*)d_in[6];
  const float* pe_w1   = (const float*)d_in[7];
  const float* pe_w2   = (const float*)d_in[8];
  float* out = (float*)d_out;

  const size_t per = 73728ull * 256ull;      // one [tokens x 256] f32 tensor
  float* q  = (float*)d_ws;
  float* k  = q + per;
  float* v  = k + per;
  float* om = v + per;
  float* t1 = q;   // q region reused for conv intermediate after attention

  const int GEMM_LDS = 2 * 128 * 264 * (int)sizeof(unsigned short);  // 135168
  const int ATTN_LDS = 29184 + 67584 + 34816;                        // 131584
  hipFuncSetAttribute((const void*)tc_qkv,
                      hipFuncAttributeMaxDynamicSharedMemorySize, GEMM_LDS);
  hipFuncSetAttribute((const void*)tc_proj,
                      hipFuncAttributeMaxDynamicSharedMemorySize, GEMM_LDS);
  hipFuncSetAttribute((const void*)tc_attn,
                      hipFuncAttributeMaxDynamicSharedMemorySize, ATTN_LDS);

  tc_qkv <<<dim3(576, 6), 256, GEMM_LDS, stream>>>(x, wq, wk, wv, q, k, v);
  tc_attn<<<dim3(576, 8), 256, ATTN_LDS, stream>>>(q, k, v, rescale, om);
  tc_proj<<<dim3(576, 2), 256, GEMM_LDS, stream>>>(om, proj_w, proj_b, out);
  tc_dwconv<<<dim3(18432), 256, 0, stream>>>(v,  pe_w1, t1, nullptr, 0);
  tc_dwconv<<<dim3(18432), 256, 0, stream>>>(t1, pe_w2, nullptr, out, 1);
}
